// TradeFlowGCN_10668698764069
// MI455X (gfx1250) — matrix operand order repaired and checked by reading, not compiled, not run on hardware
//
#include <hip/hip_runtime.h>
#include <stdint.h>

typedef __attribute__((ext_vector_type(16))) _Float16 v16h;
typedef __attribute__((ext_vector_type(8)))  float    v8f;
typedef uint32_t u32x4 __attribute__((ext_vector_type(4)));
typedef int      i32x8 __attribute__((ext_vector_type(8)));

// ---------------------------------------------------------------------------
// WMMA helpers (CDNA5 / gfx1250, wave32).  D = A(16x32 f16) x B(32x16 f16) + C(f32)
// ---------------------------------------------------------------------------
__device__ __forceinline__ v8f wmma_f16(v16h a, v16h b, v8f c) {
  // 8 args: (neg_a, A, neg_b, B, c_mod, C, reuse_a, reuse_b)
  return __builtin_amdgcn_wmma_f32_16x16x32_f16(false, a, false, b, (short)0, c,
                                                false, false);
}

union H2U { _Float16 h[2]; uint32_t u; };

// One packed element of the B-fragment layout for W[K][Nw] (row-major fp32).
// Layout = [chunk][ntile][lane][8] uint32, each uint = 2 packed f16.
// B-matrix 32x16 f16: VGPR i: lane l -> K pair {(i>=4?16:0) + (l>=16?8:0) + 2*(i%4)},
// N=l%16.  Zero-pads K beyond limit.
__device__ __forceinline__ uint32_t pack_elem(const float* __restrict__ W,
                                              int K, int Nw, int ntiles, int idx) {
  int i  = idx & 7;
  int l  = (idx >> 3) & 31;
  int jc = idx >> 8;            // c*ntiles + j
  int j  = jc % ntiles;
  int c  = jc / ntiles;
  int n  = 16 * j + (l & 15);
  int k  = 32 * c + ((i >= 4) ? 16 : 0) + ((l >> 4) << 3) + 2 * (i & 3);
  H2U t;
  t.h[0] = (k     < K) ? (_Float16)W[(size_t)k * Nw + n]       : (_Float16)0.f;
  t.h[1] = (k + 1 < K) ? (_Float16)W[(size_t)(k + 1) * Nw + n] : (_Float16)0.f;
  return t.u;
}

// Cooperative pack into LDS (used by the node kernels; weights are tiny).
__device__ inline void pack_bmat(const float* __restrict__ W, int K, int Nw,
                                 uint32_t* dst, int nchunks, int ntiles) {
  int total = nchunks * ntiles * 256;
  for (int idx = threadIdx.x; idx < total; idx += blockDim.x)
    dst[idx] = pack_elem(W, K, Nw, ntiles, idx);
}

// Load a pre-packed B fragment (8 consecutive b32 per lane -> ds_load_b128 x2).
__device__ __forceinline__ v16h bfrag(const uint32_t* p) {
  int l = threadIdx.x & 31;
  union { v16h h; uint32_t u[8]; } r;
#pragma unroll
  for (int i = 0; i < 8; ++i) r.u[i] = p[l * 8 + i];
  return r.h;
}

// A-matrix 16x32 f16 fragment from row-major f16 LDS tile (leading dim ldk).
// Lane l: M=l%16;  VGPR i holds K pair {(i>=4?16:0) + (l>=16?8:0) + 2*(i%4)}.
__device__ __forceinline__ v16h afrag_lds(const _Float16* tile, int ldk, int kbase) {
  int l = threadIdx.x & 31, m = l & 15, hi = l >> 4;
  const _Float16* row = tile + m * ldk + kbase;
  v16h a;
#pragma unroll
  for (int i = 0; i < 8; ++i) {
    int k = ((i >= 4) ? 16 : 0) + hi * 8 + 2 * (i & 3);
    a[2 * i]     = row[k];
    a[2 * i + 1] = row[k + 1];
  }
  return a;
}

// A fragment built from global fp32 rows:  a = x[node] (+ aggr[node]), f16-cvt,
// zero-padded beyond Klim.
template<int LD>
__device__ __forceinline__ v16h afrag_xsum(const float* __restrict__ x,
                                           const float* __restrict__ ag,
                                           int node0, int kbase, int Klim) {
  int l = threadIdx.x & 31, m = l & 15, hi = l >> 4;
  const float* xr = x + (size_t)(node0 + m) * LD;
  const float* ar = ag ? ag + (size_t)(node0 + m) * LD : nullptr;
  v16h a;
#pragma unroll
  for (int i = 0; i < 8; ++i) {
    int k = kbase + ((i >= 4) ? 16 : 0) + hi * 8 + 2 * (i & 3);
    float v0 = 0.f, v1 = 0.f;
    if (k < Klim)     { v0 = xr[k];     if (ar) v0 += ar[k]; }
    if (k + 1 < Klim) { v1 = xr[k + 1]; if (ar) v1 += ar[k + 1]; }
    a[2 * i]     = (_Float16)v0;
    a[2 * i + 1] = (_Float16)v1;
  }
  return a;
}

// ---------------------------------------------------------------------------
// Kernel 0: pack decoder weights (fp32 -> f16 B-fragment layout) into global
// workspace, once per launch.  Decoder blocks then TDM this blob into LDS.
// Blob: [0,9216)  = W1 (264x64 pad 288, 9 chunks x 4 ntiles)
//       [9216,10240) = W2 (64x32, 2 chunks x 2 ntiles)      -> 40,960 B total.
// ---------------------------------------------------------------------------
__global__ void __launch_bounds__(256)
pack_dec_weights_kernel(const float* __restrict__ w1, const float* __restrict__ w2,
                        uint32_t* __restrict__ wpk) {
  int idx = blockIdx.x * blockDim.x + threadIdx.x;
  if (idx < 9216)       wpk[idx] = pack_elem(w1, 264, 64, 4, idx);
  else if (idx < 10240) wpk[idx] = pack_elem(w2, 64, 32, 2, idx - 9216);
}

// ---------------------------------------------------------------------------
// Kernel 1: GINE edge message + sum-scatter.
// msg = relu(x[src] + ea @ lin_w + lin_b);  atomicAdd into aggr[dst].
// One thread per (edge, channel).  Atomic-bound; VALU is the right engine.
// ---------------------------------------------------------------------------
template<int DIN>
__global__ void __launch_bounds__(256)
edge_msg_kernel(const float* __restrict__ xin,
                const int* __restrict__ src, const int* __restrict__ dstp,
                const float* __restrict__ ea,
                const float* __restrict__ lw, const float* __restrict__ lb,
                float* __restrict__ aggr, int nE) {
  int tid = blockIdx.x * blockDim.x + threadIdx.x;
  if (tid >= nE * DIN) return;
  int c = tid & (DIN - 1);
  int e = tid / DIN;
  const float* eav = ea + (size_t)e * 8;
  float t = lb[c];
#pragma unroll
  for (int k = 0; k < 8; ++k) t += eav[k] * lw[k * DIN + c];
  float mv = xin[(size_t)src[e] * DIN + c] + t;
  mv = fmaxf(mv, 0.f);
  atomicAdd(&aggr[(size_t)dstp[e] * DIN + c], mv);
}

// ---------------------------------------------------------------------------
// Kernel 2: GINE node MLP + residual + LayerNorm, WMMA on 16-node tiles.
// h = LN( relu( relu((x+aggr)@W1+b1) @ W2 + b2 ) + res )
// Block = 64 threads = 2 waves, one 16-node tile per wave.
// ---------------------------------------------------------------------------
template<int DIN, bool L0>
__global__ void __launch_bounds__(64)
node_mlp_kernel(const float* __restrict__ xin, const float* __restrict__ aggr,
                const float* __restrict__ w1, const float* __restrict__ b1,
                const float* __restrict__ w2, const float* __restrict__ b2,
                const float* __restrict__ lng, const float* __restrict__ lnb,
                const float* __restrict__ resw, const float* __restrict__ resb,
                float* __restrict__ hout, int nNodes) {
  constexpr int C1 = (DIN + 31) / 32;                 // K-chunks for GEMM1
  __shared__ uint32_t w1p[C1 * 4 * 256];              // W1 packed B-frags
  __shared__ uint32_t w2p[2 * 4 * 256];               // W2 packed B-frags
  __shared__ uint32_t wrp[L0 ? 1024 : 8];             // residual W (layer 0)
  __shared__ _Float16 h1t[2][16 * 64];                // per-wave hidden tile
  __shared__ float    lnt[2][16 * 64];                // per-wave LN staging

  pack_bmat(w1, DIN, 64, w1p, C1, 4);
  pack_bmat(w2, 64, 64, w2p, 2, 4);
  if (L0) pack_bmat(resw, 16, 64, wrp, 1, 4);
  __syncthreads();

  int wid = threadIdx.x >> 5;
  int l = threadIdx.x & 31, m = l & 15, hi = l >> 4;
  int nTiles = (nNodes + 15) / 16;
  int tile = blockIdx.x * 2 + wid;
  if (tile >= nTiles) return;                         // wave-uniform
  int node0 = tile * 16;

  // GEMM1: (x + aggr) @ W1
  v8f acc[4] = {};
#pragma unroll
  for (int c = 0; c < C1; ++c) {
    v16h a = afrag_xsum<DIN>(xin, aggr, node0, 32 * c, DIN);
#pragma unroll
    for (int j = 0; j < 4; ++j)
      acc[j] = wmma_f16(a, bfrag(w1p + (c * 4 + j) * 256), acc[j]);
  }
  // h1 = relu(acc + b1) -> f16 LDS (row-major 16x64)
#pragma unroll
  for (int j = 0; j < 4; ++j) {
    int col = 16 * j + m;
    float bb = b1[col];
#pragma unroll
    for (int r = 0; r < 8; ++r)
      h1t[wid][(r + 8 * hi) * 64 + col] = (_Float16)fmaxf(acc[j][r] + bb, 0.f);
  }

  // Residual: layer0 = x @ Wres + bres (WMMA); else identity.
  v8f accr[4] = {};
  if (L0) {
    v16h ax = afrag_xsum<DIN>(xin, nullptr, node0, 0, DIN);
#pragma unroll
    for (int j = 0; j < 4; ++j)
      accr[j] = wmma_f16(ax, bfrag(wrp + j * 256), accr[j]);
  }

  // GEMM2: h1 @ W2
  v8f acc2[4] = {};
#pragma unroll
  for (int c = 0; c < 2; ++c) {
    v16h a = afrag_lds(h1t[wid], 64, 32 * c);
#pragma unroll
    for (int j = 0; j < 4; ++j)
      acc2[j] = wmma_f16(a, bfrag(w2p + (c * 4 + j) * 256), acc2[j]);
  }

  // relu(h2 + b2) + res  -> f32 LDS for LayerNorm
#pragma unroll
  for (int j = 0; j < 4; ++j) {
    int col = 16 * j + m;
    float bb = b2[col];
#pragma unroll
    for (int r = 0; r < 8; ++r) {
      int row = r + 8 * hi;
      float v = fmaxf(acc2[j][r] + bb, 0.f);
      float rv;
      if (L0) rv = accr[j][r] + resb[col];
      else    rv = xin[(size_t)(node0 + row) * 64 + col];
      lnt[wid][row * 64 + col] = v + rv;
    }
  }

  // LayerNorm: lanes 0..15 each own one row of 64 channels.
  if (l < 16) {
    const float* rp = &lnt[wid][l * 64];
    float mu = 0.f;
#pragma unroll
    for (int c = 0; c < 64; ++c) mu += rp[c];
    mu *= (1.f / 64.f);
    float var = 0.f;
#pragma unroll
    for (int c = 0; c < 64; ++c) { float d = rp[c] - mu; var += d * d; }
    var *= (1.f / 64.f);
    float inv = rsqrtf(var + 1e-5f);
    float* op = hout + (size_t)(node0 + l) * 64;
#pragma unroll
    for (int c = 0; c < 64; ++c)
      op[c] = (rp[c] - mu) * inv * lng[c] + lnb[c];
  }
}

// ---------------------------------------------------------------------------
// Kernel 3: edge decoder — the FLOP-dominant stage (~38 GFLOP).
// Per 16-edge tile: Z(16x264 pad 288 f16) -> W1(264x64) -> relu ->
// W2(64x32) -> relu -> w3(32x1).  36 + 4 WMMA per tile.
// Weights arrive via one Tensor-Data-Mover DMA (40,960 B) per block.
// Block = 64 threads = 2 waves; grid-stride over the 62500 tiles.
// ---------------------------------------------------------------------------
__global__ void __launch_bounds__(64)
decoder_kernel(const float* __restrict__ h, const int* __restrict__ src,
               const int* __restrict__ dstp, const float* __restrict__ ea,
               const uint32_t* __restrict__ wpk,
               const float* __restrict__ b1, const float* __restrict__ b2,
               const float* __restrict__ w3, const float* __restrict__ b3,
               float* __restrict__ out, int nE) {
  __shared__ __align__(16) uint32_t wbuf[10240]; // 40960 B: W1p (9216) + W2p (1024)
  __shared__ _Float16 zt[2][16 * 288];           // 18432 B: per-wave Z tiles
  __shared__ float    z2t[2][16 * 32];           //  4096 B: per-wave z2 staging

  // ---- TDM: DMA the packed-weight blob global -> LDS (one 1-D tile) ----
  if (threadIdx.x == 0) {
    uint64_t ga  = (uint64_t)(uintptr_t)wpk;
    uint32_t lds = (uint32_t)(uintptr_t)&wbuf[0];   // low 32 bits = LDS offset
    u32x4 g0;
    g0[0] = 1u;                                     // count=1, user descriptor
    g0[1] = lds;                                    // lds_addr
    g0[2] = (uint32_t)ga;                           // global_addr[31:0]
    g0[3] = (uint32_t)((ga >> 32) & 0x01FFFFFFu) | (2u << 30); // addr[56:32], type=2
    i32x8 g1;
    g1[0] = 0x00030000;                             // wg_mask=0, data_size=3 (8B)
    g1[1] = (int)(5120u << 16);                     // tensor_dim0[15:0]=5120
    g1[2] = (int)(1u << 16);                        // tensor_dim0[31:16]=0, dim1=1
    g1[3] = (int)(5120u << 16);                     // tile_dim0 = 5120 (x 8B)
    g1[4] = 0;                                      // tile_dim1/2 unused
    g1[5] = 5120;                                   // tensor_dim0_stride
    g1[6] = 0;
    g1[7] = 0;
    asm volatile("tensor_load_to_lds %0, %1" :: "s"(g0), "s"(g1) : "memory");
    __builtin_amdgcn_s_wait_tensorcnt(0);
  }
  __syncthreads();
  const uint32_t* w1p = wbuf;
  const uint32_t* w2p = wbuf + 9216;

  int wid = threadIdx.x >> 5;
  int l = threadIdx.x & 31, m = l & 15, hi = l >> 4;
  int nTiles = (nE + 15) / 16;
  int stride = gridDim.x * 2;
  _Float16* Z = zt[wid];

  for (int t = blockIdx.x * 2 + wid; t < nTiles; t += stride) {
    // ---- gather + build Z tile: [hs | hd | |hs-hd| | hs*hd | ea | pad] ----
    int row = m;
    int e = t * 16 + row;
    bool valid = (e < nE);
    int c0 = hi * 32;                         // two lanes per row, 32 ch each
    if (valid) {
      const float* hs = h + (size_t)src[e]  * 64;
      const float* hd = h + (size_t)dstp[e] * 64;
#pragma unroll 4
      for (int c = c0; c < c0 + 32; ++c) {
        float a = hs[c], b = hd[c];
        Z[row * 288 + c]       = (_Float16)a;
        Z[row * 288 + 64 + c]  = (_Float16)b;
        Z[row * 288 + 128 + c] = (_Float16)fabsf(a - b);
        Z[row * 288 + 192 + c] = (_Float16)(a * b);
      }
      if (hi == 0) {
#pragma unroll
        for (int k = 0; k < 8; ++k)
          Z[row * 288 + 256 + k] = (_Float16)ea[(size_t)e * 8 + k];
#pragma unroll
        for (int k = 264; k < 288; ++k) Z[row * 288 + k] = (_Float16)0.f;
      }
    } else {
      for (int c = c0; c < c0 + 32; ++c) {
        Z[row * 288 + c] = (_Float16)0.f;       Z[row * 288 + 64 + c] = (_Float16)0.f;
        Z[row * 288 + 128 + c] = (_Float16)0.f; Z[row * 288 + 192 + c] = (_Float16)0.f;
      }
      if (hi == 0)
        for (int k = 256; k < 288; ++k) Z[row * 288 + k] = (_Float16)0.f;
    }

    // gfx1250 global_prefetch of next tile's gathered rows
    int tn = t + stride;
    if (tn < nTiles && hi == 0) {
      int en = tn * 16 + row;
      if (en < nE) __builtin_prefetch(h + (size_t)src[en] * 64, 0, 0);
    }

    // ---- GEMM1: Z(16x288) @ W1(288x64), 9 K-chunks x 4 N-tiles ----
    v8f acc[4] = {};
#pragma unroll
    for (int c = 0; c < 9; ++c) {
      v16h a = afrag_lds(Z, 288, 32 * c);
#pragma unroll
      for (int j = 0; j < 4; ++j)
        acc[j] = wmma_f16(a, bfrag(w1p + (c * 4 + j) * 256), acc[j]);
    }
    // h1 = relu(acc + b1), stored back into Z region with stride 64
#pragma unroll
    for (int j = 0; j < 4; ++j) {
      int col = 16 * j + m;
      float bb = b1[col];
#pragma unroll
      for (int r = 0; r < 8; ++r)
        Z[(r + 8 * hi) * 64 + col] = (_Float16)fmaxf(acc[j][r] + bb, 0.f);
    }

    // ---- GEMM2: h1(16x64) @ W2(64x32), 2 K-chunks x 2 N-tiles ----
    v8f acc2[2] = {};
#pragma unroll
    for (int c = 0; c < 2; ++c) {
      v16h a = afrag_lds(Z, 64, 32 * c);
#pragma unroll
      for (int j = 0; j < 2; ++j)
        acc2[j] = wmma_f16(a, bfrag(w2p + (c * 2 + j) * 256), acc2[j]);
    }
#pragma unroll
    for (int j = 0; j < 2; ++j) {
      int col = 16 * j + m;
      float bb = b2[col];
#pragma unroll
      for (int r = 0; r < 8; ++r)
        z2t[wid][(r + 8 * hi) * 32 + col] = fmaxf(acc2[j][r] + bb, 0.f);
    }

    // ---- final 32 -> 1, lanes 0..15 one edge each ----
    if (l < 16) {
      int e2 = t * 16 + l;
      if (e2 < nE) {
        float s = b3[0];
#pragma unroll
        for (int c = 0; c < 32; ++c) s += z2t[wid][l * 32 + c] * w3[c];
        out[e2] = s;
      }
    }
  }
}

// ---------------------------------------------------------------------------
extern "C" void kernel_launch(void* const* d_in, const int* in_sizes, int n_in,
                              void* d_out, int out_size, void* d_ws, size_t ws_size,
                              hipStream_t stream) {
  (void)n_in; (void)out_size; (void)ws_size;
  const float* x  = (const float*)d_in[0];
  const int*   ei = (const int*)d_in[1];
  const float* ea = (const float*)d_in[2];
  const int N = in_sizes[0] / 16;
  const int E = in_sizes[1] / 2;
  const int* src  = ei;
  const int* dstp = ei + E;

  const float* l0_lin_w = (const float*)d_in[3];
  const float* l0_lin_b = (const float*)d_in[4];
  const float* l0_w1    = (const float*)d_in[5];
  const float* l0_b1    = (const float*)d_in[6];
  const float* l0_w2    = (const float*)d_in[7];
  const float* l0_b2    = (const float*)d_in[8];
  const float* l0_ln_g  = (const float*)d_in[9];
  const float* l0_ln_b  = (const float*)d_in[10];
  const float* l0_res_w = (const float*)d_in[11];
  const float* l0_res_b = (const float*)d_in[12];
  const float* l1_lin_w = (const float*)d_in[13];
  const float* l1_lin_b = (const float*)d_in[14];
  const float* l1_w1    = (const float*)d_in[15];
  const float* l1_b1    = (const float*)d_in[16];
  const float* l1_w2    = (const float*)d_in[17];
  const float* l1_b2    = (const float*)d_in[18];
  const float* l1_ln_g  = (const float*)d_in[19];
  const float* l1_ln_b  = (const float*)d_in[20];
  const float* l2_lin_w = (const float*)d_in[21];
  const float* l2_lin_b = (const float*)d_in[22];
  const float* l2_w1    = (const float*)d_in[23];
  const float* l2_b1    = (const float*)d_in[24];
  const float* l2_w2    = (const float*)d_in[25];
  const float* l2_b2    = (const float*)d_in[26];
  const float* l2_ln_g  = (const float*)d_in[27];
  const float* l2_ln_b  = (const float*)d_in[28];
  const float* dec_w1   = (const float*)d_in[29];
  const float* dec_b1   = (const float*)d_in[30];
  const float* dec_w2   = (const float*)d_in[31];
  const float* dec_b2   = (const float*)d_in[32];
  const float* dec_w3   = (const float*)d_in[33];
  const float* dec_b3   = (const float*)d_in[34];

  float* ws   = (float*)d_ws;
  float* aggr = ws;                               // N*64 f32
  float* hA   = ws + (size_t)N * 64;              // N*64 f32
  float* hB   = hA + (size_t)N * 64;              // N*64 f32
  uint32_t* wpk = (uint32_t*)(hB + (size_t)N * 64); // 10240 u32 packed dec weights
  float* out  = (float*)d_out;

  const int nTilesN = (N + 15) / 16;
  dim3 nodeGrid((nTilesN + 1) / 2), nodeBlk(64);

  // pack decoder weights once
  pack_dec_weights_kernel<<<40, 256, 0, stream>>>(dec_w1, dec_w2, wpk);

  // ---- layer 0 (D_in = 16, linear residual) ----
  hipMemsetAsync(aggr, 0, (size_t)N * 16 * sizeof(float), stream);
  edge_msg_kernel<16><<<(E * 16 + 255) / 256, 256, 0, stream>>>(
      x, src, dstp, ea, l0_lin_w, l0_lin_b, aggr, E);
  node_mlp_kernel<16, true><<<nodeGrid, nodeBlk, 0, stream>>>(
      x, aggr, l0_w1, l0_b1, l0_w2, l0_b2, l0_ln_g, l0_ln_b,
      l0_res_w, l0_res_b, hA, N);

  // ---- layer 1 (D_in = 64, identity residual) ----
  hipMemsetAsync(aggr, 0, (size_t)N * 64 * sizeof(float), stream);
  edge_msg_kernel<64><<<(E * 64 + 255) / 256, 256, 0, stream>>>(
      hA, src, dstp, ea, l1_lin_w, l1_lin_b, aggr, E);
  node_mlp_kernel<64, false><<<nodeGrid, nodeBlk, 0, stream>>>(
      hA, aggr, l1_w1, l1_b1, l1_w2, l1_b2, l1_ln_g, l1_ln_b,
      nullptr, nullptr, hB, N);

  // ---- layer 2 ----
  hipMemsetAsync(aggr, 0, (size_t)N * 64 * sizeof(float), stream);
  edge_msg_kernel<64><<<(E * 64 + 255) / 256, 256, 0, stream>>>(
      hB, src, dstp, ea, l2_lin_w, l2_lin_b, aggr, E);
  node_mlp_kernel<64, false><<<nodeGrid, nodeBlk, 0, stream>>>(
      hB, aggr, l2_w1, l2_b1, l2_w2, l2_b2, l2_ln_g, l2_ln_b,
      nullptr, nullptr, hA, N);

  // ---- decoder (dominant FLOPs: WMMA + TDM weight staging) ----
  int nTilesE = (E + 15) / 16;
  int dgrid = (nTilesE + 1) / 2;
  if (dgrid > 4096) dgrid = 4096;
  decoder_kernel<<<dgrid, 64, 0, stream>>>(
      hA, src, dstp, ea, wpk, dec_b1, dec_b2, dec_w3, dec_b3, out, E);
}
